// DeepRiskModel_78907139162346
// MI455X (gfx1250) — matrix operand order repaired and checked by reading, not compiled, not run on hardware
//
#include <hip/hip_runtime.h>
#include <hip/hip_bf16.h>
#include <cstdint>
#include <cstddef>

// ---------------- model constants (from reference) ----------------
#define T_STEPS 64
#define P_DIM   64
#define H_DIM   128
#define G3      384      // 3*H
#define EPS_BN  1e-5f

#define NT      32       // node tile per block
#define TPB     512      // 16 waves

typedef __bf16 bf16_t;
typedef __attribute__((ext_vector_type(16))) __bf16 v16bf;
typedef __attribute__((ext_vector_type(8)))  float  v8f;
typedef __attribute__((ext_vector_type(4)))  float  v4f;

// ---------------- WMMA fragment helpers (CDNA5 16x16x32 bf16) ----------------
// A-matrix 16x32 bf16: lane<16 -> row=lane, K=0..15;
// lane>=16 -> row=lane-16, K=16..31 (matches B fragment k order).
__device__ __forceinline__ v16bf load_a_lds(const float* __restrict__ s, int ld,
                                            int kbase, int lane) {
  int row = lane & 15;
  int kh  = (lane >> 4) << 4;                 // 0 or 16
  const float* p = s + row * ld + kbase + kh; // 16B-aligned chunks
  v4f q0 = *(const v4f*)(p);
  v4f q1 = *(const v4f*)(p + 4);
  v4f q2 = *(const v4f*)(p + 8);
  v4f q3 = *(const v4f*)(p + 12);
  v16bf a;
#pragma unroll
  for (int i = 0; i < 4; ++i) {
    a[i]      = (bf16_t)q0[i];
    a[4 + i]  = (bf16_t)q1[i];
    a[8 + i]  = (bf16_t)q2[i];
    a[12 + i] = (bf16_t)q3[i];
  }
  return a;
}

// B-matrix 32x16 bf16: lane<16 -> col=lane, K=0..15; lane>=16 -> col=lane-16, K=16..31.
// Weights stored row-major W[n][k] (n = output col, k contiguous) -> one contiguous
// 32B load per lane.
__device__ __forceinline__ v16bf load_b_glb(const bf16_t* __restrict__ w, int ldk,
                                            int nbase, int kbase, int lane) {
  int col = lane & 15;
  int kb  = (lane >> 4) << 4;                 // 0 or 16
  return *(const v16bf*)(w + (size_t)(nbase + col) * ldk + kbase + kb);
}

__device__ __forceinline__ v8f wmma_bf16(v16bf a, v16bf b, v8f c) {
  return __builtin_amdgcn_wmma_f32_16x16x32_bf16(false, a, false, b, (short)0, c,
                                                 false, false);
}

__device__ __forceinline__ float sigmoidf_(float x) { return 1.f / (1.f + expf(-x)); }

// ---------------- fused 2-layer GRU, persistent over T ----------------
// One (m=16-row, j=16-col) item per wave: all six gate tiles -> h_new frag.
// K-tile order is rotated by a t-dependent phase so the weight-fragment addresses
// are loop-variant: this stops LICM from hoisting ~42 B-fragments out of the
// T-loop (which caused massive scratch spill/reload in earlier builds). The sum
// over K is order-independent, so results are unchanged.
__device__ __forceinline__ v8f gru_item(const float* __restrict__ sIn, int ldin, int Kin,
                                        const bf16_t* __restrict__ Wih,
                                        const bf16_t* __restrict__ Whh,
                                        const float* __restrict__ sH,
                                        float bir, float biz, float bin,
                                        float bhr, float bhz, float bhn,
                                        int mbase, int jbase, int col, int lane,
                                        int t) {
  v8f ir = {}, iz = {}, in_ = {}, hr = {}, hz = {}, hn = {};
  int kp = t << 5;                       // phase rotation, multiple of 32
  const float* aIn = sIn + mbase * ldin;
#pragma unroll
  for (int kk = 0; kk < Kin; kk += 32) {
    int k = (kk + kp) & (Kin - 1);
    v16bf a  = load_a_lds(aIn, ldin, k, lane);
    v16bf b0 = load_b_glb(Wih, Kin, jbase,       k, lane);
    v16bf b1 = load_b_glb(Wih, Kin, 128 + jbase, k, lane);
    v16bf b2 = load_b_glb(Wih, Kin, 256 + jbase, k, lane);
    ir  = wmma_bf16(a, b0, ir);
    iz  = wmma_bf16(a, b1, iz);
    in_ = wmma_bf16(a, b2, in_);
  }
  const float* aH = sH + mbase * H_DIM;
#pragma unroll
  for (int kk = 0; kk < H_DIM; kk += 32) {
    int k = (kk + kp) & (H_DIM - 1);
    v16bf a  = load_a_lds(aH, H_DIM, k, lane);
    v16bf b0 = load_b_glb(Whh, H_DIM, jbase,       k, lane);
    v16bf b1 = load_b_glb(Whh, H_DIM, 128 + jbase, k, lane);
    v16bf b2 = load_b_glb(Whh, H_DIM, 256 + jbase, k, lane);
    hr = wmma_bf16(a, b0, hr);
    hz = wmma_bf16(a, b1, hz);
    hn = wmma_bf16(a, b2, hn);
  }
  int rbase = mbase + ((lane >> 4) << 3);
  v8f hnew;
#pragma unroll
  for (int v = 0; v < 8; ++v) {
    float hold = sH[(rbase + v) * H_DIM + col];
    float r = sigmoidf_(ir[v] + bir + hr[v] + bhr);
    float z = sigmoidf_(iz[v] + biz + hz[v] + bhz);
    float n = tanhf(in_[v] + bin + r * (hn[v] + bhn));
    hnew[v] = (1.f - z) * n + z * hold;
  }
  return hnew;
}

__device__ __forceinline__ void store_h(float* __restrict__ sH, int mbase, int col,
                                        int lane, v8f h) {
  int rbase = mbase + ((lane >> 4) << 3);
#pragma unroll
  for (int v = 0; v < 8; ++v) sH[(rbase + v) * H_DIM + col] = h[v];
}

// Block = 512 threads = 16 waves; each block owns NT=32 nodes for the whole sequence.
// Each wave owns one (m,j) item per layer.
__global__ __launch_bounds__(TPB, 1) void gru2_kernel(
    const float* __restrict__ X, int Tsteps, int N,
    const bf16_t* __restrict__ W0i, const bf16_t* __restrict__ W0h,
    const bf16_t* __restrict__ W1i, const bf16_t* __restrict__ W1h,
    const float* __restrict__ b0i, const float* __restrict__ b0h,
    const float* __restrict__ b1i, const float* __restrict__ b1h,
    float* __restrict__ out) {
  __shared__ float sX[NT * P_DIM];      // 8 KB
  __shared__ float sH0[NT * H_DIM];     // 16 KB
  __shared__ float sH1[NT * H_DIM];     // 16 KB
  __shared__ float sB[4 * G3];          // 6 KB biases

  int tid  = threadIdx.x;
  int lane = tid & 31;
  int wave = tid >> 5;
  int tile0 = blockIdx.x * NT;
  int nodeStride = Tsteps * P_DIM;

  if (tid < G3) {
    sB[tid]          = b0i[tid];
    sB[G3 + tid]     = b0h[tid];
    sB[2 * G3 + tid] = b1i[tid];
    sB[3 * G3 + tid] = b1h[tid];
  }
  for (int i = tid; i < NT * H_DIM; i += TPB) { sH0[i] = 0.f; sH1[i] = 0.f; }
  __syncthreads();

  int j  = (wave & 7) * 16;   // hidden 16-col tile (8 tiles)
  int m  = (wave >> 3) * 16;  // node 16-row tile (2 tiles)
  int col = j + (lane & 15);

  // hoist per-wave biases out of the T loop (constant over time)
  float l0_bir = sB[col],          l0_biz = sB[128 + col],          l0_bin = sB[256 + col];
  float l0_bhr = sB[G3 + col],     l0_bhz = sB[G3 + 128 + col],     l0_bhn = sB[G3 + 256 + col];
  float l1_bir = sB[2 * G3 + col], l1_biz = sB[2 * G3 + 128 + col], l1_bin = sB[2 * G3 + 256 + col];
  float l1_bhr = sB[3 * G3 + col], l1_bhz = sB[3 * G3 + 128 + col], l1_bhn = sB[3 * G3 + 256 + col];

  for (int t = 0; t < Tsteps; ++t) {
    // stage x_t tile (f32) into LDS; prefetch next step's rows into cache
    for (int i = tid; i < NT * P_DIM; i += TPB) {
      int r = i >> 6;
      int n = tile0 + r;
      int nn = n < N ? n : N - 1;
      const float* src = X + (size_t)nn * nodeStride + t * P_DIM + (i & 63);
      sX[i] = *src;
      if (t + 1 < Tsteps) __builtin_prefetch(src + P_DIM, 0, 0);
    }
    __syncthreads();
    // ---- layer 0 ----
    v8f hA = gru_item(sX, P_DIM, P_DIM, W0i, W0h, sH0,
                      l0_bir, l0_biz, l0_bin, l0_bhr, l0_bhz, l0_bhn,
                      m, j, col, lane, t);
    __syncthreads();                    // all reads of old h0 done
    store_h(sH0, m, col, lane, hA);
    __syncthreads();
    // ---- layer 1 (input = new h0) ----
    hA = gru_item(sH0, H_DIM, H_DIM, W1i, W1h, sH1,
                  l1_bir, l1_biz, l1_bin, l1_bhr, l1_bhz, l1_bhn,
                  m, j, col, lane, t);
    __syncthreads();
    store_h(sH1, m, col, lane, hA);
    __syncthreads();
  }
  for (int i = tid; i < NT * H_DIM; i += TPB) {
    int n = tile0 + (i >> 7);
    if (n < N) out[(size_t)n * H_DIM + (i & 127)] = sH1[i];
  }
}

// ---------------- small kernels ----------------
__global__ void f32_to_bf16_kernel(const float* __restrict__ src,
                                   bf16_t* __restrict__ dst, int n) {
  int i = blockIdx.x * 256 + threadIdx.x;
  if (i < n) dst[i] = (bf16_t)src[i];
}

// h = X_t @ W ; a_s = h . att_src ; a_d = h . att_dst   (one wave per node)
__global__ void gat_h_kernel(const float* __restrict__ Xseq, const float* __restrict__ W,
                             const float* __restrict__ att_s, const float* __restrict__ att_d,
                             float* __restrict__ h, float* __restrict__ a_s,
                             float* __restrict__ a_d, int N) {
  int c = threadIdx.x & 31;
  int r = threadIdx.x >> 5;  // 0..7
  int n = blockIdx.x * 8 + r;
  if (n >= N) return;
  const float* x = Xseq + (size_t)n * (T_STEPS * P_DIM) + (T_STEPS - 1) * P_DIM;
  float hv = 0.f;
#pragma unroll 8
  for (int p = 0; p < P_DIM; ++p) hv += x[p] * W[p * 32 + c];
  h[(size_t)n * 32 + c] = hv;
  float s = hv * att_s[c];
  float d = hv * att_d[c];
  for (int o = 16; o > 0; o >>= 1) {
    s += __shfl_xor(s, o, 32);
    d += __shfl_xor(d, o, 32);
  }
  if (c == 0) { a_s[n] = s; a_d[n] = d; }
}

// pass A: ex[e] = exp(leaky_relu(a_s[src]+a_d[dst])) ; denom[dst] += ex
// (softmax shift cancels in alpha = ex/denom, so no segment-max needed)
__global__ void gat_edge1_kernel(const int* __restrict__ ei, const float* __restrict__ a_s,
                                 const float* __restrict__ a_d, float* __restrict__ ex,
                                 float* __restrict__ denom, int E, int N) {
  int e = blockIdx.x * 256 + threadIdx.x;
  if (e >= E + N) return;
  int s, d;
  if (e < E) { s = ei[e]; d = ei[E + e]; } else { s = d = e - E; }
  float v = a_s[s] + a_d[d];
  v = v > 0.f ? v : 0.2f * v;
  float x = expf(v);
  ex[e] = x;
  atomicAdd(denom + d, x);
}

// pass B: accum[dst][c] += ex[e] * h[src][c]
__global__ void gat_edge2_kernel(const int* __restrict__ ei, const float* __restrict__ ex,
                                 const float* __restrict__ h, float* __restrict__ accum,
                                 int E, int N) {
  long long gid = (long long)blockIdx.x * 256 + threadIdx.x;
  int e = (int)(gid >> 5);
  if (e >= E + N) return;
  int c = (int)(gid & 31);
  int s, d;
  if (e < E) { s = ei[e]; d = ei[E + e]; } else { s = d = e - E; }
  atomicAdd(accum + (size_t)d * 32 + c, ex[e] * h[(size_t)s * 32 + c]);
}

// residual = X_t - concat(accum/denom + bias, 0-pad)
__global__ void gat_resid_kernel(const float* __restrict__ Xseq, const float* __restrict__ accum,
                                 const float* __restrict__ denom, const float* __restrict__ bias,
                                 float* __restrict__ resid, int N) {
  int gid = blockIdx.x * 256 + threadIdx.x;
  if (gid >= N * 64) return;
  int n = gid >> 6, p = gid & 63;
  float xv = Xseq[(size_t)n * (T_STEPS * P_DIM) + (T_STEPS - 1) * P_DIM + p];
  float r = xv;
  if (p < 32) r = xv - (accum[(size_t)n * 32 + p] / denom[n] + bias[p]);
  resid[gid] = r;
}

// projections + column stats for batchnorm
__global__ void proj_kernel(const float* __restrict__ h_t, const float* __restrict__ h_cs,
                            const float* __restrict__ Wt, const float* __restrict__ bt,
                            const float* __restrict__ Wc, const float* __restrict__ bc,
                            float* __restrict__ fpre, float* __restrict__ ssum,
                            float* __restrict__ ssq, int N) {
  int gid = blockIdx.x * 256 + threadIdx.x;
  if (gid >= N * 32) return;
  int n = gid >> 5, k = gid & 31;
  const float* hv;
  const float* w;
  float b;
  if (k < 16) { hv = h_t + (size_t)n * H_DIM;  w = Wt + k * H_DIM;        b = bt[k]; }
  else        { hv = h_cs + (size_t)n * H_DIM; w = Wc + (k - 16) * H_DIM; b = bc[k - 16]; }
  float acc = b;
#pragma unroll 8
  for (int i = 0; i < H_DIM; ++i) acc += hv[i] * w[i];
  fpre[gid] = acc;
  atomicAdd(ssum + k, acc);
  atomicAdd(ssq + k, acc * acc);
}

__global__ void bn_kernel(const float* __restrict__ fpre, const float* __restrict__ ssum,
                          const float* __restrict__ ssq, const float* __restrict__ g1,
                          const float* __restrict__ b1, const float* __restrict__ g2,
                          const float* __restrict__ b2, float* __restrict__ out, int N) {
  int gid = blockIdx.x * 256 + threadIdx.x;
  if (gid >= N * 32) return;
  int k = gid & 31;
  float inv_n = 1.f / (float)N;
  float m = ssum[k] * inv_n;
  float v = ssq[k] * inv_n - m * m;
  float g = k < 16 ? g1[k] : g2[k - 16];
  float bb = k < 16 ? b1[k] : b2[k - 16];
  out[gid] = (fpre[gid] - m) * rsqrtf(v + EPS_BN) * g + bb;
}

// ---------------- host launcher ----------------
static inline size_t align256(size_t x) { return (x + 255) & ~(size_t)255; }

extern "C" void kernel_launch(void* const* d_in, const int* in_sizes, int n_in,
                              void* d_out, int out_size, void* d_ws, size_t ws_size,
                              hipStream_t stream) {
  (void)n_in; (void)out_size; (void)ws_size;
  const int N = in_sizes[0] / (T_STEPS * P_DIM);
  const int E = in_sizes[1] / 2;

  const float* X   = (const float*)d_in[0];
  const int*   ei  = (const int*)d_in[1];
  const float* t_Wi0 = (const float*)d_in[2],  *t_Wh0 = (const float*)d_in[3];
  const float* t_bi0 = (const float*)d_in[4],  *t_bh0 = (const float*)d_in[5];
  const float* t_Wi1 = (const float*)d_in[6],  *t_Wh1 = (const float*)d_in[7];
  const float* t_bi1 = (const float*)d_in[8],  *t_bh1 = (const float*)d_in[9];
  const float* c_Wi0 = (const float*)d_in[10], *c_Wh0 = (const float*)d_in[11];
  const float* c_bi0 = (const float*)d_in[12], *c_bh0 = (const float*)d_in[13];
  const float* c_Wi1 = (const float*)d_in[14], *c_Wh1 = (const float*)d_in[15];
  const float* c_bi1 = (const float*)d_in[16], *c_bh1 = (const float*)d_in[17];
  const float* gat_W = (const float*)d_in[18];
  const float* att_s = (const float*)d_in[19];
  const float* att_d = (const float*)d_in[20];
  const float* gat_b = (const float*)d_in[21];
  const float* pWt = (const float*)d_in[22], *pbt = (const float*)d_in[23];
  const float* pWc = (const float*)d_in[24], *pbc = (const float*)d_in[25];
  const float* bn1g = (const float*)d_in[26], *bn1b = (const float*)d_in[27];
  const float* bn2g = (const float*)d_in[28], *bn2b = (const float*)d_in[29];

  // workspace layout
  char* ws = (char*)d_ws;
  size_t off = 0;
  float* h_t   = (float*)(ws + off); off = align256(off + (size_t)N * H_DIM * 4);
  float* h_cs  = (float*)(ws + off); off = align256(off + (size_t)N * H_DIM * 4);
  float* gat_h = (float*)(ws + off); off = align256(off + (size_t)N * 32 * 4);
  float* a_s   = (float*)(ws + off); off = align256(off + (size_t)N * 4);
  float* a_d   = (float*)(ws + off); off = align256(off + (size_t)N * 4);
  float* exb   = (float*)(ws + off); off = align256(off + (size_t)(E + N) * 4);
  float* denom = (float*)(ws + off); off = align256(off + (size_t)N * 4);
  float* accum = (float*)(ws + off); off = align256(off + (size_t)N * 32 * 4);
  float* resid = (float*)(ws + off); off = align256(off + (size_t)N * 64 * 4);
  float* fpre  = (float*)(ws + off); off = align256(off + (size_t)N * 32 * 4);
  float* ssum  = (float*)(ws + off); off = align256(off + 32 * 4);
  float* ssq   = (float*)(ws + off); off = align256(off + 32 * 4);
  bf16_t* wbf  = (bf16_t*)(ws + off); // 8 matrices, 2*172032 bf16 elems
  bf16_t* bt_Wi0 = wbf;                 // 384*64
  bf16_t* bt_Wh0 = wbf + 24576;         // 384*128
  bf16_t* bt_Wi1 = wbf + 73728;
  bf16_t* bt_Wh1 = wbf + 122880;
  bf16_t* bc_Wi0 = wbf + 172032;
  bf16_t* bc_Wh0 = wbf + 196608;
  bf16_t* bc_Wi1 = wbf + 245760;
  bf16_t* bc_Wh1 = wbf + 294912;

  // 1) convert GRU weights to bf16 (row-major == B-fragment friendly)
  {
    struct { const float* s; bf16_t* d; int n; } cv[8] = {
      { t_Wi0, bt_Wi0, G3 * P_DIM }, { t_Wh0, bt_Wh0, G3 * H_DIM },
      { t_Wi1, bt_Wi1, G3 * H_DIM }, { t_Wh1, bt_Wh1, G3 * H_DIM },
      { c_Wi0, bc_Wi0, G3 * P_DIM }, { c_Wh0, bc_Wh0, G3 * H_DIM },
      { c_Wi1, bc_Wi1, G3 * H_DIM }, { c_Wh1, bc_Wh1, G3 * H_DIM },
    };
    for (int i = 0; i < 8; ++i)
      f32_to_bf16_kernel<<<(cv[i].n + 255) / 256, 256, 0, stream>>>(cv[i].s, cv[i].d, cv[i].n);
  }

  // 2) big fused 2-layer GRU over T=64 (gru_t) -> h_t
  int gruBlocks = (N + NT - 1) / NT;
  gru2_kernel<<<gruBlocks, TPB, 0, stream>>>(X, T_STEPS, N,
                                             bt_Wi0, bt_Wh0, bt_Wi1, bt_Wh1,
                                             t_bi0, t_bh0, t_bi1, t_bh1, h_t);

  // 3) GAT: h, attention logits
  gat_h_kernel<<<(N + 7) / 8, 256, 0, stream>>>(X, gat_W, att_s, att_d, gat_h, a_s, a_d, N);

  hipMemsetAsync(denom, 0, (size_t)N * 4, stream);
  hipMemsetAsync(accum, 0, (size_t)N * 32 * 4, stream);
  hipMemsetAsync(ssum, 0, 32 * 4, stream);
  hipMemsetAsync(ssq, 0, 32 * 4, stream);

  gat_edge1_kernel<<<(E + N + 255) / 256, 256, 0, stream>>>(ei, a_s, a_d, exb, denom, E, N);
  {
    long long total = (long long)(E + N) * 32;
    gat_edge2_kernel<<<(unsigned)((total + 255) / 256), 256, 0, stream>>>(ei, exb, gat_h,
                                                                          accum, E, N);
  }
  gat_resid_kernel<<<(N * 64 + 255) / 256, 256, 0, stream>>>(X, accum, denom, gat_b, resid, N);

  // 4) gru_cs: same fused kernel with T=1 on residual (node stride = 64)
  gru2_kernel<<<gruBlocks, TPB, 0, stream>>>(resid, 1, N,
                                             bc_Wi0, bc_Wh0, bc_Wi1, bc_Wh1,
                                             c_bi0, c_bh0, c_bi1, c_bh1, h_cs);

  // 5) projections + batchnorm
  proj_kernel<<<(N * 32 + 255) / 256, 256, 0, stream>>>(h_t, h_cs, pWt, pbt, pWc, pbc,
                                                        fpre, ssum, ssq, N);
  bn_kernel<<<(N * 32 + 255) / 256, 256, 0, stream>>>(fpre, ssum, ssq, bn1g, bn1b,
                                                      bn2g, bn2b, (float*)d_out, N);
}
